// HybridQuantumKernelModel_65481071409065
// MI455X (gfx1250) — matrix operand order repaired and checked by reading, not compiled, not run on hardware
//
#include <hip/hip_runtime.h>
#include <hip/hip_bf16.h>
#include <cstdint>

typedef __attribute__((ext_vector_type(8)))  __bf16 bf16x8;
typedef __attribute__((ext_vector_type(16))) __bf16 bf16x16;
typedef __attribute__((ext_vector_type(8)))  float  f32x8;

#define BATCH 8192
#define DRAW  784      // 28*28 = 16*7*7
#define DP    800      // padded K (25 chunks of 32)
#define NKC   25
#define TS    40       // LDS bf16 tile row stride (80B, multiple of 16B)
#define TP    129      // LDS f32 transpose-tile row stride (conflict-free col reads)

// ---------------------------------------------------------------------------
// CDNA5 async global->LDS copy (16B) + ASYNCcnt wait, via inline asm.
// Generic pointer to __shared__ has the LDS offset in its low 32 bits.
// ---------------------------------------------------------------------------
__device__ __forceinline__ void async_cp16(void* ldsDst, const void* gSrc) {
  unsigned loff = (unsigned)(uintptr_t)ldsDst;
  asm volatile("global_load_async_to_lds_b128 %0, %1, off"
               :: "v"(loff), "v"(gSrc) : "memory");
}
__device__ __forceinline__ void wait_async0() {
  asm volatile("s_wait_asynccnt 0" ::: "memory");
}

// ---------------------------------------------------------------------------
// Gram + RBF kernel, triangular launch (2080 blocks cover r<=c tiles):
//   out[i,j] = exp(-gamma * max(sqn[i]+sqn[j]-2*dot(x_i,x_j),0))
// dot via bf16 hi/lo split: 4 WMMA terms -> ~f32 accuracy.
// Off-diagonal tiles also emit the mirrored block via an LDS transpose so
// both triangles get coalesced row-major stores.
// ---------------------------------------------------------------------------
__global__ __launch_bounds__(256) void gram_rbf_kernel(
    const __bf16* __restrict__ xhi, const __bf16* __restrict__ xlo,
    const float* __restrict__ sqn, const float* __restrict__ gptr,
    float* __restrict__ out)
{
  __shared__ __align__(16) unsigned char smem[81920]; // bf16 tiles, then f32 T
  __bf16* tiles = (__bf16*)smem;

  const int tid  = threadIdx.x;
  const int wave = tid >> 5;
  const int lane = tid & 31;

  // decode linear triangular index t -> (r, c) with r <= c, 64x64 tiles
  const int t = blockIdx.x;
  int r = (int)((129.0f - sqrtf(16641.0f - 8.0f * (float)t)) * 0.5f);
  if (r < 0) r = 0;
  if (r > 63) r = 63;
  while (64 * (r + 1) - ((r + 1) * r) / 2 <= t) ++r;
  while (64 * r - (r * (r - 1)) / 2 > t) --r;
  const int c = r + (t - (64 * r - (r * (r - 1)) / 2));
  const int rowBase = r * 128, colBase = c * 128;
  const float gamma = gptr[0];

  f32x8 acc[8] = {};

  auto tileptr = [&](int buf, int tt) -> __bf16* {
    return tiles + (size_t)(buf * 4 + tt) * (128 * TS);
  };

  // fill one double-buffer stage with async b128 copies (2 chunks/thread/tile)
  auto fill = [&](int buf, int kc) {
#pragma unroll
    for (int p = 0; p < 2; ++p) {
      int cidx = p * 256 + tid;              // 0..511
      int row  = cidx >> 2;                  // 0..127
      int seg  = cidx & 3;                   // 4 x 8 bf16 segments
      int loff = row * TS + seg * 8;
      size_t goffA = (size_t)(rowBase + row) * DP + (size_t)kc * 32 + seg * 8;
      size_t goffB = (size_t)(colBase + row) * DP + (size_t)kc * 32 + seg * 8;
      async_cp16(tileptr(buf, 0) + loff, xhi + goffA);
      async_cp16(tileptr(buf, 1) + loff, xlo + goffA);
      async_cp16(tileptr(buf, 2) + loff, xhi + goffB);
      async_cp16(tileptr(buf, 3) + loff, xlo + goffB);
    }
  };

  const int m  = lane & 15;                  // A row / B col / C col within 16x16
  const int kh = lane >> 4;                  // lane-half select
  const int aoff = (wave * 16 + m) * TS + kh * 8;   // A: K segs at kh*8, 16+kh*8
  const int boffBase = m * TS + kh * 8;             // B: K segs at kh*8, 16+kh*8

  fill(0, 0);
  wait_async0();
  __syncthreads();

  int buf = 0;
  for (int kc = 0; kc < NKC; ++kc) {
    if (kc + 1 < NKC) fill(buf ^ 1, kc + 1);

    const __bf16* Ah = tileptr(buf, 0);
    const __bf16* Al = tileptr(buf, 1);
    const __bf16* Bh = tileptr(buf, 2);
    const __bf16* Bl = tileptr(buf, 3);

    union U { bf16x16 v; bf16x8 h[2]; };
    U ah, al;
    ah.h[0] = *(const bf16x8*)(Ah + aoff);
    ah.h[1] = *(const bf16x8*)(Ah + aoff + 16);
    al.h[0] = *(const bf16x8*)(Al + aoff);
    al.h[1] = *(const bf16x8*)(Al + aoff + 16);

#pragma unroll
    for (int n = 0; n < 8; ++n) {
      int bo = n * 16 * TS + boffBase;
      U bh, bl;
      bh.h[0] = *(const bf16x8*)(Bh + bo);
      bh.h[1] = *(const bf16x8*)(Bh + bo + 16);
      bl.h[0] = *(const bf16x8*)(Bl + bo);
      bl.h[1] = *(const bf16x8*)(Bl + bo + 16);
      acc[n] = __builtin_amdgcn_wmma_f32_16x16x32_bf16(false, ah.v, false, bh.v,
                                                       (short)0, acc[n], false, false);
      acc[n] = __builtin_amdgcn_wmma_f32_16x16x32_bf16(false, ah.v, false, bl.v,
                                                       (short)0, acc[n], false, false);
      acc[n] = __builtin_amdgcn_wmma_f32_16x16x32_bf16(false, al.v, false, bh.v,
                                                       (short)0, acc[n], false, false);
      acc[n] = __builtin_amdgcn_wmma_f32_16x16x32_bf16(false, al.v, false, bl.v,
                                                       (short)0, acc[n], false, false);
    }

    if (kc + 1 < NKC) wait_async0();
    __syncthreads();
    buf ^= 1;
  }

  // epilogue: RBF. C/D layout: M = kh*8 + rr, N = m.
  float rsq[8];
#pragma unroll
  for (int rr = 0; rr < 8; ++rr)
    rsq[rr] = sqn[rowBase + wave * 16 + kh * 8 + rr];

  float kv[8][8];
#pragma unroll
  for (int n = 0; n < 8; ++n) {
    int col = colBase + n * 16 + m;
    float csq = sqn[col];
#pragma unroll
    for (int rr = 0; rr < 8; ++rr) {
      float d = rsq[rr] + csq - 2.f * acc[n][rr];
      d = fmaxf(d, 0.f);
      kv[n][rr] = __expf(-gamma * d);
    }
  }

  // direct (upper-triangle) block, coalesced row-major stores
#pragma unroll
  for (int n = 0; n < 8; ++n) {
    int col = colBase + n * 16 + m;
#pragma unroll
    for (int rr = 0; rr < 8; ++rr) {
      size_t row = (size_t)(rowBase + wave * 16 + kh * 8 + rr);
      out[row * BATCH + col] = kv[n][rr];
    }
  }

  // mirrored (lower-triangle) block via LDS transpose; skip on diagonal tiles
  if (r != c) {
    float* T = (float*)smem;                 // 128 x TP floats (66 KB), tiles dead
#pragma unroll
    for (int n = 0; n < 8; ++n) {
      int lc = n * 16 + m;
#pragma unroll
      for (int rr = 0; rr < 8; ++rr) {
        int lr = wave * 16 + kh * 8 + rr;
        T[lr * TP + lc] = kv[n][rr];
      }
    }
    __syncthreads();
#pragma unroll
    for (int i = 0; i < 16; ++i) {
      int lc = wave * 16 + i;
      size_t orow = (size_t)(colBase + lc) * BATCH + rowBase;
#pragma unroll
      for (int j = 0; j < 4; ++j)
        out[orow + j * 32 + lane] = T[(j * 32 + lane) * TP + lc];
    }
  }
}

// ---------------------------------------------------------------------------
// CNN side
// ---------------------------------------------------------------------------
__global__ __launch_bounds__(256) void conv1_kernel(const float* __restrict__ x,
    const float* __restrict__ w1, const float* __restrict__ b1,
    float* __restrict__ h1)
{
  __shared__ float img[784];
  __shared__ float w[72];
  __shared__ float bb[8];
  int b = blockIdx.x;
  for (int i = threadIdx.x; i < 784; i += 256) img[i] = x[(size_t)b * 784 + i];
  if (threadIdx.x < 72) w[threadIdx.x] = w1[threadIdx.x];
  if (threadIdx.x < 8)  bb[threadIdx.x] = b1[threadIdx.x];
  __syncthreads();
  for (int t = threadIdx.x; t < 1568; t += 256) {
    int c = t / 196, p = t % 196, oi = p / 14, oj = p % 14;
    float mx = 0.f;
    for (int di = 0; di < 2; ++di)
      for (int dj = 0; dj < 2; ++dj) {
        int i = 2 * oi + di, j = 2 * oj + dj;
        float acc = bb[c];
#pragma unroll
        for (int ki = 0; ki < 3; ++ki)
#pragma unroll
          for (int kj = 0; kj < 3; ++kj) {
            int y = i + ki - 1, z = j + kj - 1;
            if (y >= 0 && y < 28 && z >= 0 && z < 28)
              acc += img[y * 28 + z] * w[c * 9 + ki * 3 + kj];
          }
        mx = fmaxf(mx, fmaxf(acc, 0.f));
      }
    h1[(size_t)b * 1568 + t] = mx;
  }
}

__global__ __launch_bounds__(256) void conv2_kernel(const float* __restrict__ h1,
    const float* __restrict__ w2g, const float* __restrict__ b2g,
    float* __restrict__ h2)
{
  __shared__ float in[1568];
  __shared__ float w2[1152];
  __shared__ float b2[16];
  int b = blockIdx.x;
  for (int i = threadIdx.x; i < 1568; i += 256) in[i] = h1[(size_t)b * 1568 + i];
  for (int i = threadIdx.x; i < 1152; i += 256) w2[i] = w2g[i];
  if (threadIdx.x < 16) b2[threadIdx.x] = b2g[threadIdx.x];
  __syncthreads();
  for (int t = threadIdx.x; t < 784; t += 256) {
    int c2 = t / 49, rr = t % 49, oi = rr / 7, oj = rr % 7;
    float mx = 0.f;
    for (int di = 0; di < 2; ++di)
      for (int dj = 0; dj < 2; ++dj) {
        int i = 2 * oi + di, j = 2 * oj + dj;
        float acc = b2[c2];
        for (int ci = 0; ci < 8; ++ci)
#pragma unroll
          for (int ki = 0; ki < 3; ++ki)
#pragma unroll
            for (int kj = 0; kj < 3; ++kj) {
              int y = i + ki - 1, z = j + kj - 1;
              if (y >= 0 && y < 14 && z >= 0 && z < 14)
                acc += in[ci * 196 + y * 14 + z] * w2[c2 * 72 + ci * 9 + ki * 3 + kj];
            }
        mx = fmaxf(mx, fmaxf(acc, 0.f));
      }
    h2[(size_t)b * 784 + t] = mx;
  }
}

// transpose fc1 weights: w[64][784] -> wT[784][64]
__global__ void wt_kernel(const float* __restrict__ w, float* __restrict__ wT)
{
  int idx = blockIdx.x * 256 + threadIdx.x;   // 50176 total
  if (idx < 64 * 784) {
    int o = idx / 784, k = idx % 784;
    wT[k * 64 + o] = w[idx];
  }
}

// fc1: 32 batch rows per block, 8 rows per thread -> weights read once per 8 FMAs
__global__ __launch_bounds__(256) void fc1_kernel(const float* __restrict__ h2,
    const float* __restrict__ wT, const float* __restrict__ bias,
    float* __restrict__ o)
{
  __shared__ float v[32 * 784];               // 100 KB
  int b0 = blockIdx.x * 32;
  for (int i = threadIdx.x; i < 32 * 784; i += 256)
    v[i] = h2[(size_t)b0 * 784 + i];
  __syncthreads();
  int oi = threadIdx.x & 63, rg = threadIdx.x >> 6;   // 4 row-groups of 8
  const float* vb = &v[rg * 8 * 784];
  float acc[8];
#pragma unroll
  for (int j = 0; j < 8; ++j) acc[j] = bias[oi];
  for (int k = 0; k < 784; ++k) {
    float wk = wT[k * 64 + oi];
#pragma unroll
    for (int j = 0; j < 8; ++j) acc[j] += vb[j * 784 + k] * wk;
  }
#pragma unroll
  for (int j = 0; j < 8; ++j)
    o[(size_t)(b0 + rg * 8 + j) * 64 + oi] = fmaxf(acc[j], 0.f);
}

__global__ void fc2_kernel(const float* __restrict__ f, const float* __restrict__ w,
                           const float* __restrict__ bias, float* __restrict__ emb)
{
  int idx = blockIdx.x * 256 + threadIdx.x;   // 32768
  int b = idx >> 2, o = idx & 3;
  float acc = bias[o];
#pragma unroll
  for (int k = 0; k < 64; ++k) acc += f[(size_t)b * 64 + k] * w[o * 64 + k];
  emb[idx] = acc;
}

__global__ __launch_bounds__(1024) void bn_kernel(float* emb, const float* g,
                                                  const float* bt)
{
  __shared__ float ssum[4], ssq[4];
  int tid = threadIdx.x;
  if (tid < 4) { ssum[tid] = 0.f; ssq[tid] = 0.f; }
  __syncthreads();
  float ls[4] = {0.f, 0.f, 0.f, 0.f}, lq[4] = {0.f, 0.f, 0.f, 0.f};
  for (int rr = tid; rr < 8192; rr += 1024)
#pragma unroll
    for (int c = 0; c < 4; ++c) {
      float v = emb[rr * 4 + c];
      ls[c] += v; lq[c] += v * v;
    }
#pragma unroll
  for (int c = 0; c < 4; ++c) { atomicAdd(&ssum[c], ls[c]); atomicAdd(&ssq[c], lq[c]); }
  __syncthreads();
  float mu[4], isd[4];
#pragma unroll
  for (int c = 0; c < 4; ++c) {
    mu[c] = ssum[c] * (1.f / 8192.f);
    float var = ssq[c] * (1.f / 8192.f) - mu[c] * mu[c];
    isd[c] = rsqrtf(var + 1e-5f);
  }
  for (int i = tid; i < 32768; i += 1024) {
    int c = i & 3;
    emb[i] = (emb[i] - mu[c]) * isd[c] * g[c] + bt[c];
  }
}

__global__ void cast_split_kernel(const float* __restrict__ x,
                                  __bf16* __restrict__ xhi, __bf16* __restrict__ xlo)
{
  size_t idx = (size_t)blockIdx.x * 256 + threadIdx.x;   // 8192*800 total
  size_t rr = idx / DP, c = idx % DP;
  float v = (c < DRAW) ? x[rr * DRAW + c] : 0.f;
  __bf16 hi = (__bf16)v;
  float lo = v - (float)hi;
  xhi[idx] = hi;
  xlo[idx] = (__bf16)lo;
}

__global__ void sqn_kernel(const float* __restrict__ x, float* __restrict__ sqn)
{
  int warp = threadIdx.x >> 5, lane = threadIdx.x & 31;
  int row = blockIdx.x * 8 + warp;
  float s = 0.f;
  for (int c = lane; c < DRAW; c += 32) { float v = x[(size_t)row * DRAW + c]; s += v * v; }
#pragma unroll
  for (int o = 16; o > 0; o >>= 1) s += __shfl_xor(s, o, 32);
  if (lane == 0) sqn[row] = s;
}

__global__ void zero_kernel(float* q) { q[blockIdx.x * 1024 + threadIdx.x] = 0.f; }

// ---------------------------------------------------------------------------
extern "C" void kernel_launch(void* const* d_in, const int* in_sizes, int n_in,
                              void* d_out, int out_size, void* d_ws, size_t ws_size,
                              hipStream_t stream)
{
  const float* x   = (const float*)d_in[0];
  const float* c1w = (const float*)d_in[1];
  const float* c1b = (const float*)d_in[2];
  const float* c2w = (const float*)d_in[3];
  const float* c2b = (const float*)d_in[4];
  const float* f1w = (const float*)d_in[5];
  const float* f1b = (const float*)d_in[6];
  const float* f2w = (const float*)d_in[7];
  const float* f2b = (const float*)d_in[8];
  const float* bng = (const float*)d_in[9];
  const float* bnb = (const float*)d_in[10];
  const float* gam = (const float*)d_in[11];
  (void)in_sizes; (void)n_in; (void)out_size; (void)ws_size;

  char* ws = (char*)d_ws;
  size_t off = 0;
  auto alloc = [&](size_t bytes) {
    void* p = ws + off;
    off = (off + bytes + 255) & ~(size_t)255;
    return p;
  };
  float*  h1  = (float*)alloc(sizeof(float) * (size_t)BATCH * 8 * 14 * 14);
  float*  h2  = (float*)alloc(sizeof(float) * (size_t)BATCH * DRAW);
  float*  f1o = (float*)alloc(sizeof(float) * (size_t)BATCH * 64);
  float*  wT  = (float*)alloc(sizeof(float) * 64 * DRAW);
  __bf16* xhi = (__bf16*)alloc(sizeof(__bf16) * (size_t)BATCH * DP);
  __bf16* xlo = (__bf16*)alloc(sizeof(__bf16) * (size_t)BATCH * DP);
  float*  sqn = (float*)alloc(sizeof(float) * BATCH);

  float* classical = (float*)d_out;
  float* kfeat     = classical + (size_t)BATCH * 4;
  float* quantum   = kfeat + (size_t)BATCH * BATCH;

  conv1_kernel<<<BATCH, 256, 0, stream>>>(x, c1w, c1b, h1);
  conv2_kernel<<<BATCH, 256, 0, stream>>>(h1, c2w, c2b, h2);
  wt_kernel<<<196, 256, 0, stream>>>(f1w, wT);
  fc1_kernel<<<BATCH / 32, 256, 0, stream>>>(h2, wT, f1b, f1o);
  fc2_kernel<<<128, 256, 0, stream>>>(f1o, f2w, f2b, classical);
  bn_kernel<<<1, 1024, 0, stream>>>(classical, bng, bnb);
  cast_split_kernel<<<(BATCH * DP) / 256, 256, 0, stream>>>(x, xhi, xlo);
  sqn_kernel<<<BATCH / 8, 256, 0, stream>>>(x, sqn);
  zero_kernel<<<32, 1024, 0, stream>>>(quantum);
  gram_rbf_kernel<<<2080, 256, 0, stream>>>(xhi, xlo, sqn, gam, kfeat);
}
